// CosFormerAttention_42030549959122
// MI455X (gfx1250) — matrix operand order repaired and checked by reading, not compiled, not run on hardware
//
#include <hip/hip_runtime.h>

typedef __attribute__((ext_vector_type(16))) _Float16 v16h;
typedef __attribute__((ext_vector_type(8)))  _Float16 h8;
typedef __attribute__((ext_vector_type(4)))  _Float16 h4;
typedef __attribute__((ext_vector_type(8)))  float    v8f;

constexpr int BB = 4, LL = 1024, HH = 8, DD = 64;
constexpr float TEMP_INV = 0.125f;   // 1 / TEMPERATURE(8.0)
constexpr float EPSN     = 1e-8f;

// 16-bit A-fragment K map: elements 0..15 -> K = g*16 + e
// (two contiguous 16B runs).
__device__ __forceinline__ v16h load_a_frag(const _Float16* row, int base, int g) {
  h8 lo = *reinterpret_cast<const h8*>(row + base + g * 16);
  h8 hi = *reinterpret_cast<const h8*>(row + base + g * 16 + 8);
  v16h a;
#pragma unroll
  for (int e = 0; e < 8; ++e) { a[e] = lo[e]; a[e + 8] = hi[e]; }
  return a;
}

__global__ __launch_bounds__(256)
void cosformer_attn_kernel(const float* __restrict__ Q,
                           const float* __restrict__ K,
                           const float* __restrict__ V,
                           float* __restrict__ Out,
                           float* __restrict__ Attn) {
  // Logits -> exp -> probabilities in place, f16. Row stride 1032 halfs
  // (2064B = 16B-aligned, kills the 16-way bank conflict on column reads).
  __shared__ _Float16 sP[16][1032];
  __shared__ _Float16 sQ[16][72];      // normalized Q tile, f16, padded
  __shared__ float    sRed[4][32][8];  // PV partial-C exchange

  const int tid  = threadIdx.x;
  const int wid  = tid >> 5;     // wave 0..7
  const int lane = tid & 31;
  const int ln   = lane & 15;    // M (A/C) or N (B/C) index
  const int g    = lane >> 4;    // lane-group

  const int it = blockIdx.x & 63;   // 16-row tile index within L
  const int bh = blockIdx.x >> 6;   // b*H + h
  const int bb = bh >> 3;
  const int hh = bh & 7;

  // ---------- Stage 1: L2-normalize Q tile (16x64) into LDS as f16 ----------
  {
    const int r  = tid >> 4;   // row 0..15
    const int cg = tid & 15;   // 4 columns per thread
    const float* qrow = Q + ((size_t)(bb * LL + it * 16 + r) * HH + hh) * DD;
    float4 qv = *reinterpret_cast<const float4*>(qrow + cg * 4);
    float ss = qv.x * qv.x + qv.y * qv.y + qv.z * qv.z + qv.w * qv.w;
    ss += __shfl_xor(ss, 1);
    ss += __shfl_xor(ss, 2);
    ss += __shfl_xor(ss, 4);
    ss += __shfl_xor(ss, 8);   // reduction stays inside the 16-lane row group
    float inv = 1.0f / fmaxf(sqrtf(ss), EPSN);
    h4 qh;
    qh[0] = (_Float16)(qv.x * inv);
    qh[1] = (_Float16)(qv.y * inv);
    qh[2] = (_Float16)(qv.z * inv);
    qh[3] = (_Float16)(qv.w * inv);
    *reinterpret_cast<h4*>(&sQ[r][cg * 4]) = qh;   // one ds_store_b64
  }
  __syncthreads();

  // Hoisted A fragments of Qn (invariant over all 64 column tiles).
  const v16h aq0 = load_a_frag(&sQ[ln][0], 0, g);   // d = K
  const v16h aq1 = load_a_frag(&sQ[ln][0], 32, g);  // d = 32 + K

  // ---------- Stage 2: S = Qn @ Kn^T * (1/T), 8 column tiles per wave ------
  for (int itj = 0; itj < 8; ++itj) {
    const int jt = itj * 8 + wid;
    const float* krow = K + ((size_t)(bb * LL + jt * 16 + ln) * HH + hh) * DD;
    if (itj < 7) {  // prefetch next tile's row (global_prefetch_b8 path)
      const float* knext =
          K + ((size_t)(bb * LL + (jt + 8) * 16 + ln) * HH + hh) * DD;
      __builtin_prefetch(knext, 0, 1);
    }

    // Lane (n,g) needs d in [g*16, g*16+16) and [32+g*16, 32+g*16+16)
    float kv[32];
#pragma unroll
    for (int q4 = 0; q4 < 4; ++q4) {
      float4 t = *reinterpret_cast<const float4*>(krow + g * 16 + q4 * 4);
      kv[q4 * 4 + 0] = t.x; kv[q4 * 4 + 1] = t.y;
      kv[q4 * 4 + 2] = t.z; kv[q4 * 4 + 3] = t.w;
    }
#pragma unroll
    for (int q4 = 0; q4 < 4; ++q4) {
      float4 t = *reinterpret_cast<const float4*>(krow + 32 + g * 16 + q4 * 4);
      kv[16 + q4 * 4 + 0] = t.x; kv[16 + q4 * 4 + 1] = t.y;
      kv[16 + q4 * 4 + 2] = t.z; kv[16 + q4 * 4 + 3] = t.w;
    }
    float own = 0.f;
#pragma unroll
    for (int i = 0; i < 32; ++i) own += kv[i] * kv[i];
    float tot = own + __shfl_xor(own, 16);  // lanes n / n+16 cover all 64 d's
    float inv = 1.0f / fmaxf(sqrtf(tot), EPSN);

    v16h bk0, bk1;
#pragma unroll
    for (int e = 0; e < 16; ++e) {
      bk0[e] = (_Float16)(kv[e] * inv);        // B frag: K = g*16+e, d = K
      bk1[e] = (_Float16)(kv[16 + e] * inv);   // B frag: d = 32 + K
    }
    v8f c = {};
    c = __builtin_amdgcn_wmma_f32_16x16x32_f16(false, aq0, false, bk0,
                                               (short)0, c, false, false);
    c = __builtin_amdgcn_wmma_f32_16x16x32_f16(false, aq1, false, bk1,
                                               (short)0, c, false, false);
    // C layout: lane n = ln, VGPR r -> row r + g*8 (scatter b16 stores)
#pragma unroll
    for (int r = 0; r < 8; ++r)
      sP[r + g * 8][jt * 16 + ln] = (_Float16)(c[r] * TEMP_INV);
  }
  __syncthreads();

  // ---------- Stage 3: softmax rows; write attn (f32), keep P (f16) --------
  // Lane owns 4 consecutive columns per step: cols = lane*4 + i*128.
#pragma unroll
  for (int rr2 = 0; rr2 < 2; ++rr2) {
    const int rr = wid * 2 + rr2;
    _Float16* prow = &sP[rr][0];

    float m = -3.0e38f;
#pragma unroll
    for (int i = 0; i < 8; ++i) {
      h4 x = *reinterpret_cast<const h4*>(prow + lane * 4 + i * 128);
      m = fmaxf(m, fmaxf(fmaxf((float)x[0], (float)x[1]),
                         fmaxf((float)x[2], (float)x[3])));
    }
    m = fmaxf(m, __shfl_xor(m, 16));
    m = fmaxf(m, __shfl_xor(m, 8));
    m = fmaxf(m, __shfl_xor(m, 4));
    m = fmaxf(m, __shfl_xor(m, 2));
    m = fmaxf(m, __shfl_xor(m, 1));

    float s = 0.f;
#pragma unroll
    for (int i = 0; i < 8; ++i) {
      h4 x = *reinterpret_cast<h4*>(prow + lane * 4 + i * 128);
      float e0 = __expf((float)x[0] - m);
      float e1 = __expf((float)x[1] - m);
      float e2 = __expf((float)x[2] - m);
      float e3 = __expf((float)x[3] - m);
      s += (e0 + e1) + (e2 + e3);
      h4 eh;   // exp in [exp(-0.25),1] -> safe in f16; reused in pass 3
      eh[0] = (_Float16)e0; eh[1] = (_Float16)e1;
      eh[2] = (_Float16)e2; eh[3] = (_Float16)e3;
      *reinterpret_cast<h4*>(prow + lane * 4 + i * 128) = eh;
    }
    s += __shfl_xor(s, 16);
    s += __shfl_xor(s, 8);
    s += __shfl_xor(s, 4);
    s += __shfl_xor(s, 2);
    s += __shfl_xor(s, 1);
    const float invs = 1.0f / s;

    float* arow = Attn + ((size_t)bh * LL + it * 16 + rr) * LL;
#pragma unroll
    for (int i = 0; i < 8; ++i) {
      h4 eh = *reinterpret_cast<h4*>(prow + lane * 4 + i * 128);
      float4 p;
      p.x = (float)eh[0] * invs;  p.y = (float)eh[1] * invs;
      p.z = (float)eh[2] * invs;  p.w = (float)eh[3] * invs;
      // lane-contiguous: one global_store_b128 per lane, 512B/wave burst
      *reinterpret_cast<float4*>(arow + lane * 4 + i * 128) = p;
      h4 ph;
      ph[0] = (_Float16)p.x; ph[1] = (_Float16)p.y;
      ph[2] = (_Float16)p.z; ph[3] = (_Float16)p.w;
      *reinterpret_cast<h4*>(prow + lane * 4 + i * 128) = ph;
    }
  }
  __syncthreads();

  // ---------- Stage 4: Out = P @ V  (4 D-tiles x 2 K-halves across waves) --
  const int nt    = wid & 3;   // D tile: columns nt*16 .. nt*16+15
  const int khalf = wid >> 2;  // half of the K=1024 contraction
  v8f acc = {};
  for (int c = 0; c < 16; ++c) {
    const int kb = (khalf * 16 + c) * 32;
    v16h ap = load_a_frag(&sP[ln][0], kb, g);   // two ds_load_b128
    v16h bv;
    const float* vcol = V + ((size_t)(bb * LL + kb + g * 16) * HH + hh) * DD
                          + nt * 16 + ln;
#pragma unroll
    for (int e = 0; e < 16; ++e)    // B frag: row = kb + g*16 + e, col = d
      bv[e] = (_Float16)vcol[(size_t)e * HH * DD];
    acc = __builtin_amdgcn_wmma_f32_16x16x32_f16(false, ap, false, bv,
                                                 (short)0, acc, false, false);
  }
  if (khalf == 1) {
#pragma unroll
    for (int r = 0; r < 8; ++r) sRed[nt][lane][r] = acc[r];
  }
  __syncthreads();
  if (khalf == 0) {
#pragma unroll
    for (int r = 0; r < 8; ++r) {
      float o = acc[r] + sRed[nt][lane][r];
      Out[((size_t)bh * LL + it * 16 + r + g * 8) * DD + nt * 16 + ln] = o;
    }
  }
}

extern "C" void kernel_launch(void* const* d_in, const int* in_sizes, int n_in,
                              void* d_out, int out_size, void* d_ws, size_t ws_size,
                              hipStream_t stream) {
  const float* q = (const float*)d_in[0];
  const float* k = (const float*)d_in[1];
  const float* v = (const float*)d_in[2];
  float* out  = (float*)d_out;
  float* attn = out + (size_t)BB * HH * LL * DD;  // (out, attn) concatenated

  dim3 grid(BB * HH * (LL / 16));  // 2048 blocks
  dim3 block(256);                 // 8 wave32 waves
  hipLaunchKernelGGL(cosformer_attn_kernel, grid, block, 0, stream,
                     q, k, v, out, attn);
}
